// HyperAttention_78297253806432
// MI455X (gfx1250) — matrix operand order repaired and checked by reading, not compiled, not run on hardware
//
#include <hip/hip_runtime.h>
#include <hip/hip_bf16.h>
#include <cstdint>
#include <cstddef>

// ---------------------------------------------------------------------------
// HyperAttention for MI455X (gfx1250, wave32, WMMA bf16 16x16x32, f32 accum)
//
// B=4096, V=25, DIM=384, H=6, HD=64.  SCALE = 1/8.
// Pipeline: cvt(x,w) -> GEMM q -> GEMM kv -> per-(b,h) WMMA attention
//           -> GEMM proj (+bias, f32 out)
// ---------------------------------------------------------------------------

typedef __bf16 bf16_t;
typedef bf16_t v16bf __attribute__((ext_vector_type(16)));
typedef float  v8f   __attribute__((ext_vector_type(8)));

#define BATCH 4096
#define NV    25
#define DIMC  384
#define NH    6
#define HDIM  64
#define MROWS (BATCH * NV)   // 102400

__device__ __forceinline__ bf16_t f2bf(float x) { return (bf16_t)x; }

__device__ __forceinline__ v8f wmma_bf16(v16bf a, v16bf b, v8f c) {
  // v_wmma_f32_16x16x32_bf16  D = A x B + C   (f32 accumulate)
  return __builtin_amdgcn_wmma_f32_16x16x32_bf16(false, a, false, b,
                                                 (short)0, c, false, false);
}

// A-operand: lane holds 16 contiguous K values starting at (lane>>4)*16
__device__ __forceinline__ v16bf load_fragA(const bf16_t* p) {
  union { struct { uint4 lo, hi; } s; v16bf v; } u;
  u.s.lo = *(const uint4*)(p);
  u.s.hi = *(const uint4*)(p + 8);
  return u.v;
}
// B-operand: lane holds 16 contiguous K values starting at (lane>>4)*16
__device__ __forceinline__ v16bf load_fragB(const bf16_t* p) {
  union { struct { uint4 lo, hi; } s; v16bf v; } u;
  u.s.lo = *(const uint4*)(p);
  u.s.hi = *(const uint4*)(p + 8);
  return u.v;
}
__device__ __forceinline__ int fragA_k(int half, int t) {
  return (half << 4) + t;
}
__device__ __forceinline__ int fragB_k(int half, int t) {
  return (half << 4) + t;
}

// ---------------------------------------------------------------------------
// f32 -> bf16 conversion (n divisible by 4)
// ---------------------------------------------------------------------------
__global__ void cvt_kernel(const float* __restrict__ in, bf16_t* __restrict__ out,
                           int n4) {
  int i = blockIdx.x * blockDim.x + threadIdx.x;
  if (i < n4) {
    float4 v = ((const float4*)in)[i];
    union { bf16_t e[4]; uint2 u; } o;
    o.e[0] = f2bf(v.x); o.e[1] = f2bf(v.y);
    o.e[2] = f2bf(v.z); o.e[3] = f2bf(v.w);
    ((uint2*)out)[i] = o.u;
  }
}

// ---------------------------------------------------------------------------
// Generic GEMM: C[M,N] = A[M,384] * W[N,384]^T
// Block tile 128x128, 8 waves as 4(M) x 2(N); each wave 32x64 = 2x4 WMMA tiles.
// Double-buffered LDS staging of 128x32 A/B panels. M,N multiples of 128.
// ---------------------------------------------------------------------------
template <bool OUTF32>
__global__ __launch_bounds__(256)
void gemm_kernel(const bf16_t* __restrict__ A, const bf16_t* __restrict__ Bw,
                 void* __restrict__ Cout, const float* __restrict__ bias,
                 int N) {
  constexpr int K = 384;
  constexpr int KITER = K / 32;
  __shared__ bf16_t As[2][128 * 32];
  __shared__ bf16_t Bs[2][128 * 32];

  const int tid  = threadIdx.x;
  const int lane = tid & 31;
  const int wave = tid >> 5;
  const int l16  = lane & 15;
  const int half = lane >> 4;

  const int rowBlk = blockIdx.y * 128;
  const int colBlk = blockIdx.x * 128;
  const int wm = (wave >> 1) * 32;   // wave M offset: 0/32/64/96
  const int wn = (wave & 1) * 64;    // wave N offset: 0/64

  // staging: 2 threads per row, 16 elements (32B) each
  const int str = tid >> 1;
  const int ss  = (tid & 1) * 16;
  const bf16_t* gA = A  + (size_t)(rowBlk + str) * K + ss;
  const bf16_t* gB = Bw + (size_t)(colBlk + str) * K + ss;

  auto loadTile = [&](int buf, int k0) {
    uint4 a0 = *(const uint4*)(gA + k0);
    uint4 a1 = *(const uint4*)(gA + k0 + 8);
    uint4 b0 = *(const uint4*)(gB + k0);
    uint4 b1 = *(const uint4*)(gB + k0 + 8);
    *(uint4*)&As[buf][str * 32 + ss]     = a0;
    *(uint4*)&As[buf][str * 32 + ss + 8] = a1;
    *(uint4*)&Bs[buf][str * 32 + ss]     = b0;
    *(uint4*)&Bs[buf][str * 32 + ss + 8] = b1;
  };

  const v8f zero8 = {0.f, 0.f, 0.f, 0.f, 0.f, 0.f, 0.f, 0.f};
  v8f acc[2][4];
#pragma unroll
  for (int i = 0; i < 2; ++i)
#pragma unroll
    for (int j = 0; j < 4; ++j) acc[i][j] = zero8;

  loadTile(0, 0);
  __syncthreads();

  for (int kk = 0; kk < KITER; ++kk) {
    const int buf = kk & 1;
    if (kk + 1 < KITER) loadTile(buf ^ 1, (kk + 1) * 32);
    if (kk + 2 < KITER) {  // -> global_prefetch_b8
      __builtin_prefetch(gA + (kk + 2) * 32, 0, 1);
      __builtin_prefetch(gB + (kk + 2) * 32, 0, 1);
    }
    v16bf af[2], bfg[4];
#pragma unroll
    for (int mt = 0; mt < 2; ++mt)
      af[mt] = load_fragA(&As[buf][(wm + mt * 16 + l16) * 32 + half * 16]);
#pragma unroll
    for (int nt = 0; nt < 4; ++nt)
      bfg[nt] = load_fragB(&Bs[buf][(wn + nt * 16 + l16) * 32 + half * 16]);
#pragma unroll
    for (int mt = 0; mt < 2; ++mt)
#pragma unroll
      for (int nt = 0; nt < 4; ++nt)
        acc[mt][nt] = wmma_bf16(af[mt], bfg[nt], acc[mt][nt]);
    __syncthreads();
  }

  // D layout: VGPR r -> M = r + 8*(lane/16), N = lane%16
#pragma unroll
  for (int mt = 0; mt < 2; ++mt) {
#pragma unroll
    for (int nt = 0; nt < 4; ++nt) {
      const int col = colBlk + wn + nt * 16 + l16;
#pragma unroll
      for (int r = 0; r < 8; ++r) {
        const int row = rowBlk + wm + mt * 16 + half * 8 + r;
        if constexpr (OUTF32) {
          ((float*)Cout)[(size_t)row * N + col] = acc[mt][nt][r] + bias[col];
        } else {
          ((bf16_t*)Cout)[(size_t)row * N + col] = f2bf(acc[mt][nt][r]);
        }
      }
    }
  }
}

// ---------------------------------------------------------------------------
// Per-(b,h) attention, one wave32 per block.
//   q[b,h,m,d]  = Qflat[b][h*1600 + d*25 + m]        (bf16, from GEMM)
//   e_k[b,h,n,d]= e[b flat][h*1600 + d*25 + n]       (f32 input, cvt on fly)
//   v[b,h,n,d]  = KVflat[b][9600 + h*1600 + d*25 + n]
// Scores S(26x25): row 25 of A = w1[h,:] -> learned bias row via same WMMA.
// attn = softmax((S[m]+S[25]) * 1/8); O = attn @ v ->
//   F[b*25+m, h*64+d] (row-major bf16) for the final projection.
// ---------------------------------------------------------------------------
__global__ __launch_bounds__(32)
void attn_kernel(const bf16_t* __restrict__ qraw, const bf16_t* __restrict__ kvraw,
                 const float* __restrict__ ein, const float* __restrict__ w1,
                 bf16_t* __restrict__ fmat) {
  __shared__ float  Sl[26][33];
  __shared__ bf16_t At[32][32];

  const int lane = threadIdx.x;
  const int l16 = lane & 15, half = lane >> 4;
  const int bh = blockIdx.x;
  const int b = bh / NH, h = bh % NH;
  const int hb = h * (HDIM * NV);  // h*1600

  const bf16_t* qb  = qraw  + (size_t)b * (NV * DIMC);       // 9600 bf16
  const bf16_t* kvb = kvraw + (size_t)b * (NV * 2 * DIMC);   // 19200 bf16
  const float*  eb  = ein   + (size_t)b * (NV * DIMC);

  const v8f zero8 = {0.f, 0.f, 0.f, 0.f, 0.f, 0.f, 0.f, 0.f};

  // ---- A = [q rows 0..24 ; w1 row 25 ; zeros], 32x64 ----
  v16bf aq[2][2], bk[2][2];
#pragma unroll
  for (int mt = 0; mt < 2; ++mt) {
    const int row = mt * 16 + l16;
#pragma unroll
    for (int kc = 0; kc < 2; ++kc) {
      v16bf f;
#pragma unroll
      for (int t = 0; t < 16; ++t) {
        const int K = kc * 32 + fragA_k(half, t);
        bf16_t v;
        if (row < 25)       v = qb[hb + K * 25 + row];
        else if (row == 25) v = f2bf(w1[h * HDIM + K]);
        else                v = f2bf(0.f);
        f[t] = v;
      }
      aq[mt][kc] = f;
    }
  }
  // ---- B = e_k (N = key token n, K = d) ----
#pragma unroll
  for (int nt = 0; nt < 2; ++nt) {
    const int n = nt * 16 + l16;
#pragma unroll
    for (int kc = 0; kc < 2; ++kc) {
      v16bf f;
#pragma unroll
      for (int t = 0; t < 16; ++t) {
        const int K = kc * 32 + fragB_k(half, t);
        f[t] = (n < 25) ? f2bf(eb[hb + K * 25 + n]) : f2bf(0.f);
      }
      bk[nt][kc] = f;
    }
  }
  v8f S[2][2];
#pragma unroll
  for (int mt = 0; mt < 2; ++mt)
#pragma unroll
    for (int nt = 0; nt < 2; ++nt) {
      v8f a = zero8;
      a = wmma_bf16(aq[mt][0], bk[nt][0], a);
      a = wmma_bf16(aq[mt][1], bk[nt][1], a);
      S[mt][nt] = a;
    }

  // dump rows 0..25 of S to LDS
#pragma unroll
  for (int mt = 0; mt < 2; ++mt)
#pragma unroll
    for (int nt = 0; nt < 2; ++nt)
#pragma unroll
      for (int r = 0; r < 8; ++r) {
        const int m = mt * 16 + half * 8 + r;
        const int n = nt * 16 + l16;
        if (m < 26) Sl[m][n] = S[mt][nt][r];
      }
  __syncthreads();

  // ---- softmax: lane = query row m; bias row = Sl[25][*] ----
  if (lane < 25) {
    float mx = -3.0e38f;
#pragma unroll
    for (int n = 0; n < 25; ++n) {
      float s = (Sl[lane][n] + Sl[25][n]) * 0.125f;
      Sl[lane][n] = s;
      mx = fmaxf(mx, s);
    }
    float sum = 0.f;
#pragma unroll
    for (int n = 0; n < 25; ++n) {
      float p = __expf(Sl[lane][n] - mx);
      Sl[lane][n] = p;
      sum += p;
    }
    const float inv = 1.0f / sum;
#pragma unroll
    for (int n = 0; n < 25; ++n) At[lane][n] = f2bf(Sl[lane][n] * inv);
#pragma unroll
    for (int n = 25; n < 32; ++n) At[lane][n] = f2bf(0.f);
  } else {
#pragma unroll
    for (int n = 0; n < 32; ++n) At[lane][n] = f2bf(0.f);
  }
  __syncthreads();

  // ---- O = attn(25x25 pad 32) @ v(25x64) ----
  v16bf aA[2];
#pragma unroll
  for (int mt = 0; mt < 2; ++mt)
    aA[mt] = load_fragA(&At[mt * 16 + l16][half * 16]);

  v16bf bV[4];
#pragma unroll
  for (int dt = 0; dt < 4; ++dt) {
    const int d = dt * 16 + l16;
    v16bf f;
#pragma unroll
    for (int t = 0; t < 16; ++t) {
      const int n = fragB_k(half, t);
      f[t] = (n < 25) ? kvb[NV * DIMC + hb + d * 25 + n] : f2bf(0.f);
    }
    bV[dt] = f;
  }
  v8f O[2][4];
#pragma unroll
  for (int mt = 0; mt < 2; ++mt)
#pragma unroll
    for (int dt = 0; dt < 4; ++dt)
      O[mt][dt] = wmma_bf16(aA[mt], bV[dt], zero8);

  // store unscrambled: F[b*25+m][h*64+d]
#pragma unroll
  for (int mt = 0; mt < 2; ++mt)
#pragma unroll
    for (int dt = 0; dt < 4; ++dt)
#pragma unroll
      for (int r = 0; r < 8; ++r) {
        const int m = mt * 16 + half * 8 + r;
        if (m < 25)
          fmat[((size_t)b * NV + m) * DIMC + h * HDIM + dt * 16 + l16] =
              f2bf(O[mt][dt][r]);
      }
}

// ---------------------------------------------------------------------------
extern "C" void kernel_launch(void* const* d_in, const int* in_sizes, int n_in,
                              void* d_out, int out_size, void* d_ws, size_t ws_size,
                              hipStream_t stream) {
  (void)in_sizes; (void)n_in; (void)out_size; (void)ws_size;
  const float* x      = (const float*)d_in[0];
  const float* e      = (const float*)d_in[1];
  const float* w1     = (const float*)d_in[2];
  const float* kv_w   = (const float*)d_in[3];
  const float* q_w    = (const float*)d_in[4];
  const float* proj_w = (const float*)d_in[5];
  const float* proj_b = (const float*)d_in[6];
  float* outp = (float*)d_out;

  char* ws = (char*)d_ws;
  size_t off = 0;
  auto wsalloc = [&](size_t bytes) -> void* {
    void* p = ws + off;
    off += (bytes + 255) & ~(size_t)255;
    return p;
  };
  bf16_t* xb    = (bf16_t*)wsalloc((size_t)MROWS * DIMC * 2);
  bf16_t* wqb   = (bf16_t*)wsalloc((size_t)DIMC * DIMC * 2);
  bf16_t* wkvb  = (bf16_t*)wsalloc((size_t)2 * DIMC * DIMC * 2);
  bf16_t* wpjb  = (bf16_t*)wsalloc((size_t)DIMC * DIMC * 2);
  bf16_t* qraw  = (bf16_t*)wsalloc((size_t)MROWS * DIMC * 2);
  bf16_t* kvraw = (bf16_t*)wsalloc((size_t)MROWS * 2 * DIMC * 2);
  bf16_t* fmat  = (bf16_t*)wsalloc((size_t)MROWS * DIMC * 2);

  auto cvt = [&](const float* src, bf16_t* dst, long n) {
    int n4 = (int)(n / 4);
    cvt_kernel<<<(n4 + 255) / 256, 256, 0, stream>>>(src, dst, n4);
  };
  cvt(x, xb, (long)MROWS * DIMC);
  cvt(q_w, wqb, (long)DIMC * DIMC);
  cvt(kv_w, wkvb, (long)2 * DIMC * DIMC);
  cvt(proj_w, wpjb, (long)DIMC * DIMC);

  // Q = x @ q_w^T   [102400 x 384]
  gemm_kernel<false><<<dim3(DIMC / 128, MROWS / 128), 256, 0, stream>>>(
      xb, wqb, qraw, nullptr, DIMC);
  // KV = x @ kv_w^T [102400 x 768]
  gemm_kernel<false><<<dim3(2 * DIMC / 128, MROWS / 128), 256, 0, stream>>>(
      xb, wkvb, kvraw, nullptr, 2 * DIMC);
  // fused attention per (b,h)
  attn_kernel<<<BATCH * NH, 32, 0, stream>>>(qraw, kvraw, e, w1, fmat);
  // out = F @ proj_w^T + proj_b  (f32)
  gemm_kernel<true><<<dim3(DIMC / 128, MROWS / 128), 256, 0, stream>>>(
      fmat, wpjb, outp, proj_b, DIMC);
}
